// FaceAttention_81578608820750
// MI455X (gfx1250) — compile-verified, matrix operand order checked by reading, not run on hardware
//
#include <hip/hip_runtime.h>

// ---------------- constants ----------------
#define BATCH   4096
#define NFACE   15
#define DIM     2048        // K
#define HID     2048        // N
#define NTILE   256
#define KSTEP   32          // one bf16 WMMA k-step
#define KITERS  (DIM / KSTEP)   // 64

#define ASZ     10240       // 128 rows * 80 B (16 B row pad)
#define BSZ     16384       // 16 nt * 1 KB fragment chunk

#define HAS_ASYNC_LDS __has_builtin(__builtin_amdgcn_global_load_async_to_lds_b128)

typedef __attribute__((ext_vector_type(16))) __bf16 v16bf;
typedef __attribute__((ext_vector_type(8)))  float  v8f;

#if HAS_ASYNC_LDS
typedef int v4i __attribute__((ext_vector_type(4)));
typedef __attribute__((address_space(1))) v4i* g_v4i_p;   // global int4*
typedef __attribute__((address_space(3))) v4i* l_v4i_p;   // LDS int4*
#endif

union Frag { uint4 q[2]; v16bf v; };

// f32 pair -> packed bf16 dword (RNE). Prefer v_cvt_pk_bf16_f32 when available.
__device__ __forceinline__ unsigned pack_bf16(float lo, float hi) {
#if __has_builtin(__builtin_amdgcn_cvt_pk_bf16_f32)
    auto r = __builtin_amdgcn_cvt_pk_bf16_f32(lo, hi);
    return __builtin_bit_cast(unsigned, r);
#else
    unsigned u0 = __float_as_uint(lo), u1 = __float_as_uint(hi);
    unsigned r0 = u0 + 0x7FFFu + ((u0 >> 16) & 1u);
    unsigned r1 = u1 + 0x7FFFu + ((u1 >> 16) & 1u);
    return __builtin_amdgcn_perm(r1, r0, 0x07060302u);   // {r1.hi16, r0.hi16}
#endif
}

template<int N>
__device__ __forceinline__ void wait_async() {
#if HAS_ASYNC_LDS
# if __has_builtin(__builtin_amdgcn_s_wait_asynccnt)
    __builtin_amdgcn_s_wait_asynccnt(N);
# else
    asm volatile("s_wait_asynccnt %0" :: "n"(N) : "memory");
# endif
#endif
}

// ---------------------------------------------------------------------------
// Kernel 1: reformat W1 (f32 row-major [K=2048][N=2048]) into bf16 fragments.
// Layout: [kt(64)][nt(128)][lane(32)][16 bf16]  (32 B per lane, contiguous).
//   lane<16 : n = nt*16 + lane,    K(i) = kt*32 + (i<8 ? i : i+8)
//   lane>=16: n = nt*16 + lane-16, K(i) = kt*32 + 8 + (i<8 ? i : i+8)
// ---------------------------------------------------------------------------
__global__ __launch_bounds__(256) void reformat_w1(const float* __restrict__ W1,
                                                   unsigned short* __restrict__ W1F) {
    int t    = blockIdx.x * 256 + threadIdx.x;      // 0 .. 64*128*32-1
    int lane = t & 31;
    int nt   = (t >> 5) & 127;
    int kt   = t >> 12;

    int n     = nt * 16 + (lane & 15);
    int kbase = kt * 32 + ((lane < 16) ? 0 : 8);

    float v[16];
#pragma unroll
    for (int i = 0; i < 16; ++i) {
        int k = kbase + (i < 8 ? i : i + 8);
        v[i] = W1[(size_t)k * HID + n];
    }
    union { unsigned u[8]; uint4 q[2]; } vals;
#pragma unroll
    for (int i = 0; i < 8; ++i) vals.u[i] = pack_bf16(v[2 * i], v[2 * i + 1]);
    uint4* dst = (uint4*)(W1F + (size_t)t * 16);
    dst[0] = vals.q[0];
    dst[1] = vals.q[1];
}

// ---------------------------------------------------------------------------
// Kernel 2: fused GEMM (relu(X@W1+b1)) + masked face-sum pooling -> Hbar[B,H]
// 512 threads = 16 waves, tile M=128 (8 batches x 16 padded faces), N=256.
// Waves 4(M) x 4(N): each wave 32x64 = 2x4 WMMAs/k-step (acc = 64 VGPRs).
// LDS double-buffered; B chunks copied with GLOBAL_LOAD_ASYNC_TO_LDS_B128
// (raw bf16 bytes, pre-formatted), A converted f32->bf16 through registers.
// One barrier per k-step.
// ---------------------------------------------------------------------------
__global__ __launch_bounds__(512) void gemm_pool(const float* __restrict__ X,
                                                 const unsigned short* __restrict__ W1F,
                                                 const float* __restrict__ b1,
                                                 const int* __restrict__ nFaces,
                                                 float* __restrict__ Hbar) {
    const int tid   = threadIdx.x;
    const int lane  = tid & 31;
    const int wid   = tid >> 5;     // 0..15
    const int waveM = wid & 3;      // rows waveM*32
    const int waveN = wid >> 2;     // cols waveN*64
    const int nBlk  = blockIdx.x;   // cols nBlk*256
    const int mBlk  = blockIdx.y;   // batches mBlk*8

    __shared__ __align__(16) unsigned char smem[2 * ASZ + 2 * BSZ];
    unsigned char* As0 = smem;              // + buf*ASZ
    unsigned char* Bs0 = smem + 2 * ASZ;    // + buf*BSZ

    // ---- A global-load mapping (4 threads per padded row, 8 f32 each) ----
    const int arow   = tid >> 2;            // 0..127
    const int face   = arow & 15;           // 15 == padding row
    const int batchA = mBlk * 8 + (arow >> 4);
    const int kpart  = (tid & 3) * 8;       // element offset within KSTEP
    const float* aSrc = (face < NFACE)
        ? (X + ((size_t)(batchA * NFACE + face)) * DIM + kpart) : nullptr;

    const unsigned char* w1fB = (const unsigned char*)W1F;
    const int ntBase = nBlk * 16;

    float4 aR[2];
#if !HAS_ASYNC_LDS
    uint4 bR[2];
#endif

    auto loadA = [&](int kt) {
        if (aSrc) {
            const float4* p = (const float4*)(aSrc + (size_t)kt * KSTEP);
            aR[0] = p[0]; aR[1] = p[1];
        } else {
            aR[0] = aR[1] = make_float4(0.f, 0.f, 0.f, 0.f);
        }
    };

    auto fetchB = [&](int kt, int buf) {
        const unsigned char* src = w1fB + ((size_t)(kt * 128 + ntBase)) * 1024
                                        + (size_t)tid * 32;
#if HAS_ASYNC_LDS
        unsigned char* dst = Bs0 + buf * BSZ + tid * 32;
        __builtin_amdgcn_global_load_async_to_lds_b128(
            (g_v4i_p)(src), (l_v4i_p)(dst), 0, 0);
        __builtin_amdgcn_global_load_async_to_lds_b128(
            (g_v4i_p)(src + 16), (l_v4i_p)(dst + 16), 0, 0);
#else
        const uint4* bp = (const uint4*)src;
        bR[0] = bp[0]; bR[1] = bp[1];
#endif
    };

    auto storeA = [&](int buf) {
        union { unsigned u[4]; uint4 q; } pk;
        const float* f = (const float*)aR;
#pragma unroll
        for (int i = 0; i < 4; ++i) pk.u[i] = pack_bf16(f[2 * i], f[2 * i + 1]);
        *(uint4*)(As0 + buf * ASZ + arow * 80 + kpart * 2) = pk.q;
#if !HAS_ASYNC_LDS
        uint4* bd = (uint4*)(Bs0 + buf * BSZ + (size_t)tid * 32);
        bd[0] = bR[0]; bd[1] = bR[1];
#endif
    };

    v8f acc[2][4];
#pragma unroll
    for (int t = 0; t < 2; ++t)
#pragma unroll
        for (int j = 0; j < 4; ++j) acc[t][j] = (v8f)(0.f);

    const int halfSel  = (lane < 16) ? 0 : 16;          // +8 k-elements, high half
    const int aRowOff  = (waveM * 32 + (lane & 15)) * 80 + halfSel;
    const int bWaveOff = (waveN * 4) * 1024 + lane * 32;

    loadA(0);
    fetchB(0, 0);
    for (int kt = 0; kt < KITERS; ++kt) {
        const int cur = kt & 1;
        storeA(cur);                               // A(kt) regs -> LDS (+B in fallback)
        if (kt + 1 < KITERS) {
            loadA(kt + 1);                         // overlap next A with compute
            fetchB(kt + 1, cur ^ 1);               // async B into other buffer
            wait_async<2>();                       // B(kt) landed; B(kt+1) in flight
        } else {
            wait_async<0>();
        }
        __syncthreads();                           // publish As[cur], Bs[cur]

        const unsigned char* aP = As0 + cur * ASZ + aRowOff;
        const unsigned char* bP = Bs0 + cur * BSZ + bWaveOff;

        Frag a0, a1;
        a0.q[0] = *(const uint4*)(aP);
        a0.q[1] = *(const uint4*)(aP + 32);
        a1.q[0] = *(const uint4*)(aP + 16 * 80);
        a1.q[1] = *(const uint4*)(aP + 16 * 80 + 32);

#pragma unroll
        for (int j = 0; j < 4; ++j) {
            Frag b;
            const unsigned char* bp = bP + (size_t)j * 1024;
            b.q[0] = *(const uint4*)(bp);
            b.q[1] = *(const uint4*)(bp + 16);
            acc[0][j] = __builtin_amdgcn_wmma_f32_16x16x32_bf16(
                false, a0.v, false, b.v, (short)0, acc[0][j], false, false);
            acc[1][j] = __builtin_amdgcn_wmma_f32_16x16x32_bf16(
                false, a1.v, false, b.v, (short)0, acc[1][j], false, false);
        }
    }

    // ---- epilogue: bias + relu + face mask + sum over 16 rows (one batch) ----
    const int nbase   = nBlk * 256 + waveN * 64;
    const int rowbase = (lane < 16) ? 0 : 8;
#pragma unroll
    for (int t = 0; t < 2; ++t) {
        const int batchE = mBlk * 8 + waveM * 2 + t;
        const int nb     = min(nFaces[batchE], NFACE);
#pragma unroll
        for (int j = 0; j < 4; ++j) {
            const int col  = nbase + j * 16 + (lane & 15);
            const float bi = b1[col];
            float s = 0.f;
#pragma unroll
            for (int r = 0; r < 8; ++r) {
                float v = acc[t][j][r] + bi;
                v = v > 0.f ? v : 0.f;
                s += ((rowbase + r) < nb) ? v : 0.f;
            }
            s += __shfl_xor(s, 16);                 // combine lane halves (wave32)
            if (lane < 16) Hbar[(size_t)batchE * HID + col] = s;
        }
    }
}

// ---------------------------------------------------------------------------
// Kernel 3: out[b,:] = (n>0) ? Hbar[b,:] @ W2 / n + b2 : 0     (tiny, BW-bound)
// ---------------------------------------------------------------------------
__global__ __launch_bounds__(256) void proj_pool(const float* __restrict__ Hbar,
                                                 const float* __restrict__ W2,
                                                 const float* __restrict__ b2,
                                                 const int* __restrict__ nFaces,
                                                 float* __restrict__ out) {
    const int b    = blockIdx.x;
    const int tid  = threadIdx.x;
    const int lane = tid & 31;
    const int wid  = tid >> 5;

    float p0 = 0.f, p1 = 0.f, p2 = 0.f;
#pragma unroll
    for (int i = 0; i < HID / 256; ++i) {
        int h   = i * 256 + tid;
        float v = Hbar[(size_t)b * HID + h];
        p0 += v * W2[h * 3 + 0];
        p1 += v * W2[h * 3 + 1];
        p2 += v * W2[h * 3 + 2];
    }
#pragma unroll
    for (int o = 16; o > 0; o >>= 1) {
        p0 += __shfl_down(p0, o);
        p1 += __shfl_down(p1, o);
        p2 += __shfl_down(p2, o);
    }
    __shared__ float red[8][3];
    if (lane == 0) { red[wid][0] = p0; red[wid][1] = p1; red[wid][2] = p2; }
    __syncthreads();
    if (tid == 0) {
        float s0 = 0.f, s1 = 0.f, s2 = 0.f;
#pragma unroll
        for (int w = 0; w < 8; ++w) { s0 += red[w][0]; s1 += red[w][1]; s2 += red[w][2]; }
        int n = min(nFaces[b], NFACE);
        if (n > 0) {
            float inv = 1.f / (float)n;
            out[b * 3 + 0] = s0 * inv + b2[0];
            out[b * 3 + 1] = s1 * inv + b2[1];
            out[b * 3 + 2] = s2 * inv + b2[2];
        } else {
            out[b * 3 + 0] = 0.f;
            out[b * 3 + 1] = 0.f;
            out[b * 3 + 2] = 0.f;
        }
    }
}

// ---------------------------------------------------------------------------
extern "C" void kernel_launch(void* const* d_in, const int* in_sizes, int n_in,
                              void* d_out, int out_size, void* d_ws, size_t ws_size,
                              hipStream_t stream) {
    const float* X   = (const float*)d_in[0];   // [4096,15,2048]
    const float* W1  = (const float*)d_in[1];   // [2048,2048]
    const float* b1  = (const float*)d_in[2];   // [2048]
    const float* W2  = (const float*)d_in[3];   // [2048,3]
    const float* b2v = (const float*)d_in[4];   // [3]
    const int*   nF  = (const int*)d_in[5];     // [4096]
    // d_in[6] = labels (unused by the math)
    float* out = (float*)d_out;                 // [4096,3]

    // workspace: 8 MB bf16-fragment W1  +  32 MB Hbar
    unsigned short* W1F = (unsigned short*)d_ws;
    float* Hbar = (float*)((char*)d_ws + (size_t)8 * 1024 * 1024);

    reformat_w1<<<1024, 256, 0, stream>>>(W1, W1F);
    gemm_pool<<<dim3(HID / NTILE, BATCH / 8), 512, 0, stream>>>(X, W1F, b1, nF, Hbar);
    proj_pool<<<BATCH, 256, 0, stream>>>(Hbar, W2, b2v, nF, out);
}
